// Kernel_Layer_16690242912928
// MI455X (gfx1250) — compile-verified, matrix operand order checked by reading, not run on hardware
//
#include <hip/hip_runtime.h>
#include <stdint.h>

// ---------------------------------------------------------------------------
// RBF similarity: out[b,p] = exp(-sqrt(||x_b||^2 + ||p_p||^2 - 2 x_b.p_p) / 2)
// B = P = 8192, F = 256, SIGMA = 1.
//
// MI455X plan: output store (268 MB) bounds the op at ~11.5us @ 23.3 TB/s.
// Cross term runs on the matrix pipe: bf16 WMMA with f32 accumulate, exact
// f32 norms, fused sqrt/exp epilogue. K-chunks are DMA'd into LDS with
// gfx1250 GLOBAL_LOAD_ASYNC_TO_LDS_B128 (ASYNCcnt), double-buffered so the
// next chunk's HBM fetch overlaps the current chunk's WMMAs.
// ---------------------------------------------------------------------------

typedef __attribute__((ext_vector_type(16))) __bf16 v16bf;
typedef __attribute__((ext_vector_type(8)))  float  v8f;

#define FDIM 256
#define BM 128
#define BN 128
#define KC 64                    // K-chunk (elements) staged in LDS
#define LDS_STRIDE 144           // bytes/row: KC*2 data + 16 pad (kills bank conflicts)
#define NCHUNK (FDIM / KC)       // 4
#define BUF_BYTES ((BM + BN) * LDS_STRIDE)

__device__ __forceinline__ unsigned short f32_to_bf16_rne(float f) {
    union { float f; uint32_t u; } c; c.f = f;
    uint32_t u = c.u;
    uint32_t r = u + 0x7FFFu + ((u >> 16) & 1u);   // round to nearest even
    return (unsigned short)(r >> 16);
}

// 64B contiguous global -> LDS async DMA (4 x b128). INST_OFFSET is added to
// BOTH the LDS and the global address (ISA 08_async_tensor 4.4), so one base
// pair covers all four beats. Tracked by ASYNCcnt, no destination VGPRs.
__device__ __forceinline__ void async_copy_64B(unsigned lds_addr,
                                               const unsigned short* g) {
    asm volatile(
        "global_load_async_to_lds_b128 %0, %1, off offset:0\n\t"
        "global_load_async_to_lds_b128 %0, %1, off offset:16\n\t"
        "global_load_async_to_lds_b128 %0, %1, off offset:32\n\t"
        "global_load_async_to_lds_b128 %0, %1, off offset:48"
        :: "v"(lds_addr), "v"(g) : "memory");
}

// Prepass: src f32 [rows, 256] -> dst16 bf16 [rows, 256], dstsq[row] = ||row||^2
__global__ __launch_bounds__(256)
void convert_and_norms(const float* __restrict__ src,
                       unsigned short* __restrict__ dst16,
                       float* __restrict__ dstsq) {
    __shared__ float red[256];
    const int row = blockIdx.x;
    const int tid = threadIdx.x;
    const float v = src[(size_t)row * FDIM + tid];
    dst16[(size_t)row * FDIM + tid] = f32_to_bf16_rne(v);
    red[tid] = v * v;
    __syncthreads();
    #pragma unroll
    for (int s = 128; s > 0; s >>= 1) {
        if (tid < s) red[tid] += red[tid + s];
        __syncthreads();
    }
    if (tid == 0) dstsq[row] = red[0];
}

// Main kernel: one 128x128 output tile per workgroup (8 wave32).
// Wave grid 2(M) x 4(N); each wave owns 64x32 = 4x2 tiles of 16x16.
__global__ __launch_bounds__(256)
void rbf_wmma(const unsigned short* __restrict__ xb,
              const unsigned short* __restrict__ pb,
              const float* __restrict__ xsq,
              const float* __restrict__ psq,
              float* __restrict__ out,
              int Bdim, int Pdim) {
    // Double-buffered LDS: [buf][ A(128 rows) | B(128 rows) ], 72 KB total.
    __shared__ __align__(16) unsigned char lds[2][BUF_BYTES];

    const int tid  = threadIdx.x;
    const int wave = tid >> 5;
    const int lane = tid & 31;
    const int wm   = wave & 1;        // wave position along M (0..1)
    const int wn   = wave >> 1;       // wave position along N (0..3)
    const int r    = lane & 15;       // matrix row (A: M, B: N) held by this lane
    const int h    = lane >> 4;       // half-wave: selects K byte-quarters

    const int blockN = blockIdx.x;    // over P (cols)
    const int blockM = blockIdx.y;    // over B (rows)

    // global->LDS DMA mapping: each thread moves 64B of one row chunk per matrix
    const int loadRow  = tid >> 1;    // 0..127
    const int loadPart = tid & 1;     // two 64B halves of the 128B row chunk

    v8f c[4][2];
    #pragma unroll
    for (int i = 0; i < 4; ++i)
        #pragma unroll
        for (int n = 0; n < 2; ++n)
            c[i][n] = (v8f){0.f,0.f,0.f,0.f,0.f,0.f,0.f,0.f};

    const unsigned short* gA = xb + (size_t)(blockM * BM + loadRow) * FDIM;
    const unsigned short* gB = pb + (size_t)(blockN * BN + loadRow) * FDIM;

    // Low 32 bits of a flat shared address are the LDS byte offset
    // (aperture base occupies only ADDR[63:32] per ISA flat addressing).
    const unsigned ldsAddr0 = (unsigned)(uintptr_t)&lds[0][0];
    const unsigned ldsAddr1 = (unsigned)(uintptr_t)&lds[1][0];

    union Frag { uint4 u[2]; v16bf v; };

    // Prologue: DMA chunk 0 into buffer 0 (8 async b128 per thread per chunk).
    {
        const unsigned la = ldsAddr0 + loadRow * LDS_STRIDE + loadPart * 64;
        async_copy_64B(la, gA + loadPart * 32);
        async_copy_64B(la + BM * LDS_STRIDE, gB + loadPart * 32);
    }

    #pragma unroll
    for (int kc = 0; kc < NCHUNK; ++kc) {
        const int buf = kc & 1;

        // Issue next chunk's DMA into the other buffer, then wait only for
        // the current chunk (8 newer async ops may remain in flight).
        if (kc + 1 < NCHUNK) {
            const unsigned la = (buf ? ldsAddr0 : ldsAddr1) +
                                loadRow * LDS_STRIDE + loadPart * 64;
            const int eoff = (kc + 1) * KC + loadPart * 32;
            async_copy_64B(la, gA + eoff);
            async_copy_64B(la + BM * LDS_STRIDE, gB + eoff);
            asm volatile("s_wait_asynccnt 0x8" ::: "memory");
        } else {
            asm volatile("s_wait_asynccnt 0x0" ::: "memory");
        }
        __syncthreads();   // chunk kc visible to all waves

        const unsigned char* bufA = &lds[buf][0];
        const unsigned char* bufB = bufA + BM * LDS_STRIDE;

        // ---- two K=32 WMMA steps per chunk ----
        #pragma unroll
        for (int kk = 0; kk < KC / 32; ++kk) {
            // ISA 16-bit A layout: lane<16 holds K0..7 (bytes 0..15) and
            // K16..23 (bytes 32..47); lane>=16 holds K8..15 and K24..31.
            const int kb = kk * 64 + h * 16;
            Frag a[4], b[2];
            #pragma unroll
            for (int i = 0; i < 4; ++i) {
                const unsigned char* p =
                    bufA + (wm * 64 + i * 16 + r) * LDS_STRIDE + kb;
                a[i].u[0] = *(const uint4*)p;
                a[i].u[1] = *(const uint4*)(p + 32);
            }
            #pragma unroll
            for (int n = 0; n < 2; ++n) {
                const unsigned char* p =
                    bufB + (wn * 32 + n * 16 + r) * LDS_STRIDE + kb;
                b[n].u[0] = *(const uint4*)p;
                b[n].u[1] = *(const uint4*)(p + 32);
            }
            #pragma unroll
            for (int i = 0; i < 4; ++i)
                #pragma unroll
                for (int n = 0; n < 2; ++n)
                    c[i][n] = __builtin_amdgcn_wmma_f32_16x16x32_bf16(
                        false, a[i].v, false, b[n].v,
                        (short)0, c[i][n], false, false);
        }
        __syncthreads();   // readers done before this buffer is re-filled
    }

    // ---- fused epilogue: d2 -> sqrt -> exp2, then store ----
    // C/D layout: VGPR j -> row j (lanes 0-15) / j+8 (lanes 16-31), col = r.
    const float kcoef = 0.7213475204444817f;   // log2(e) / (2*sigma^2)
    #pragma unroll
    for (int i = 0; i < 4; ++i) {
        const int rowBase = blockM * BM + wm * 64 + i * 16 + h * 8;
        #pragma unroll
        for (int n = 0; n < 2; ++n) {
            const int col = blockN * BN + wn * 32 + n * 16 + r;
            const float ps = psq[col];
            #pragma unroll
            for (int j = 0; j < 8; ++j) {
                const int row = rowBase + j;
                float d2 = xsq[row] + ps - 2.0f * c[i][n][j];
                d2 = fmaxf(d2, 0.0f);
                const float dist = sqrtf(d2);
                out[(size_t)row * Pdim + col] = exp2f(-dist * kcoef);
            }
        }
    }
}

extern "C" void kernel_launch(void* const* d_in, const int* in_sizes, int n_in,
                              void* d_out, int out_size, void* d_ws, size_t ws_size,
                              hipStream_t stream) {
    const float* x = (const float*)d_in[0];
    const float* p = (const float*)d_in[1];
    const int B = in_sizes[0] / FDIM;   // 8192
    const int P = in_sizes[1] / FDIM;   // 8192
    float* out = (float*)d_out;

    // Workspace layout: xb16 | pb16 | xsq | psq  (~8.5 MB total)
    unsigned char* ws = (unsigned char*)d_ws;
    unsigned short* xb  = (unsigned short*)ws;
    unsigned short* pb  = (unsigned short*)(ws + (size_t)B * FDIM * 2);
    float*          xsq = (float*)(ws + (size_t)(B + P) * FDIM * 2);
    float*          psq = xsq + B;

    convert_and_norms<<<B, 256, 0, stream>>>(x, xb, xsq);
    convert_and_norms<<<P, 256, 0, stream>>>(p, pb, psq);

    dim3 grid(P / BN, B / BM);
    rbf_wmma<<<grid, 256, 0, stream>>>(xb, pb, xsq, psq, out, B, P);
}
